// EquivariantResidualLayer_55963423867394
// MI455X (gfx1250) — compile-verified, matrix operand order checked by reading, not run on hardware
//
#include <hip/hip_runtime.h>
#include <hip/hip_bf16.h>
#include <math.h>

// Problem sizes (fixed by the reference)
#define NN    50000
#define EE    800000
#define HH    128
#define VV    32
#define EDD   64
#define EIN   320   // 2H+ED
#define EHID  320
#define EOUT  320
#define NIN   160   // H+V
#define NHID  288
#define NOUT  160

typedef __bf16 bf16;
typedef __attribute__((ext_vector_type(16))) __bf16 v16bf;
typedef __attribute__((ext_vector_type(8)))  __bf16 bf16x8;
typedef __attribute__((ext_vector_type(8)))  float  v8f;

union V16U { v16bf v; bf16x8 h[2]; };

__device__ __forceinline__ float sigmoidf_(float x) { return 1.f / (1.f + __expf(-x)); }

__device__ __forceinline__ float wave_sum(float v) {
    for (int m = 16; m >= 1; m >>= 1) v += __shfl_xor(v, m, 32);
    return v;
}

// CDNA5 async copy: global -> LDS, 16B per lane, tracked with ASYNCcnt.
// VDST = LDS byte address (wave-relative low 32 bits), VADDR = 64-bit address.
__device__ __forceinline__ void async_b128(void* lds_dst, const void* gsrc) {
    unsigned lds_off = (unsigned)(size_t)lds_dst;
    unsigned long long ga = (unsigned long long)(size_t)gsrc;
    asm volatile("global_load_async_to_lds_b128 %0, %1, off"
                 :: "v"(lds_off), "v"(ga) : "memory");
}
__device__ __forceinline__ void wait_async0() {
    asm volatile("s_wait_asynccnt 0x0" ::: "memory");
}

// Hoist the (layer-invariant) A fragments for all K-tiles into registers once.
template <int KT>
__device__ __forceinline__ void load_afrag(const bf16* __restrict__ ar, int half,
                                           V16U* __restrict__ a) {
#pragma unroll
    for (int kt = 0; kt < KT; ++kt) {
        int k0 = kt * 32 + 8 * half;
        a[kt].h[0] = *(const bf16x8*)(ar + k0);
        a[kt].h[1] = *(const bf16x8*)(ar + k0 + 16);
    }
}

// Two output tiles per K-loop sharing the register-resident A fragments:
// two independent WMMA chains (1 wave/SIMD at this LDS footprint) and only
// 2 LDS b128 loads per WMMA.
template <int KT>
__device__ __forceinline__ void wmma_pair(const V16U* __restrict__ a,
                                          const bf16* __restrict__ br0,
                                          const bf16* __restrict__ br1,
                                          int half, v8f& acc0, v8f& acc1) {
#pragma unroll
    for (int kt = 0; kt < KT; ++kt) {
        int k0 = kt * 32 + 8 * half;
        V16U b0, b1;
        b0.h[0] = *(const bf16x8*)(br0 + k0);
        b0.h[1] = *(const bf16x8*)(br0 + k0 + 16);
        b1.h[0] = *(const bf16x8*)(br1 + k0);
        b1.h[1] = *(const bf16x8*)(br1 + k0 + 16);
        acc0 = __builtin_amdgcn_wmma_f32_16x16x32_bf16(false, a[kt].v, false, b0.v,
                                                       (short)0, acc0, false, false);
        acc1 = __builtin_amdgcn_wmma_f32_16x16x32_bf16(false, a[kt].v, false, b1.v,
                                                       (short)0, acc1, false, false);
    }
}

// ---------------------------------------------------------------- zero scratch
__global__ void k_zero(float* p, long n) {
    for (long i = (long)blockIdx.x * blockDim.x + threadIdx.x; i < n;
         i += (long)gridDim.x * blockDim.x)
        p[i] = 0.f;
}

// ------------------------------------------------- layernorm + cast to bf16
__global__ void k_ln_cast(const float* __restrict__ x, const float* __restrict__ g,
                          const float* __restrict__ b, bf16* __restrict__ out) {
    int wave = threadIdx.x >> 5, lane = threadIdx.x & 31;
    int row = blockIdx.x * 8 + wave;
    if (row >= NN) return;
    const float* xr = x + (long)row * HH;
    float v0[4]; float s = 0.f;
    for (int j = 0; j < 4; j++) { v0[j] = xr[lane + 32 * j]; s += v0[j]; }
    s = wave_sum(s);
    float m = s * (1.f / HH);
    float var = 0.f;
    for (int j = 0; j < 4; j++) { float d = v0[j] - m; var += d * d; }
    var = wave_sum(var) * (1.f / HH);
    float inv = rsqrtf(var + 1e-5f);
    bf16* orow = out + (long)row * HH;
    for (int j = 0; j < 4; j++) {
        int f = lane + 32 * j;
        orow[f] = (bf16)((v0[j] - m) * inv * g[f] + b[f]);
    }
}

// ------------------------------------ transpose + cast weights to bf16 [out][in]
__global__ void k_convert(const float* __restrict__ ew1, const float* __restrict__ ew2,
                          const float* __restrict__ nw1, const float* __restrict__ nw2,
                          bf16* __restrict__ w1t, bf16* __restrict__ w2t,
                          bf16* __restrict__ nw1t, bf16* __restrict__ nw2t) {
    int tid = blockIdx.x * blockDim.x + threadIdx.x;
    int stride = gridDim.x * blockDim.x;
    for (int i = tid; i < EHID * EIN;  i += stride) { int n = i / EIN,  k = i % EIN;  w1t[i]  = (bf16)ew1[k * EHID + n]; }
    for (int i = tid; i < EOUT * EHID; i += stride) { int n = i / EHID, k = i % EHID; w2t[i]  = (bf16)ew2[k * EOUT + n]; }
    for (int i = tid; i < NHID * NIN;  i += stride) { int n = i / NIN,  k = i % NIN;  nw1t[i] = (bf16)nw1[k * NHID + n]; }
    for (int i = tid; i < NOUT * NHID; i += stride) { int n = i / NHID, k = i % NHID; nw2t[i] = (bf16)nw2[k * NOUT + n]; }
}

// -------------------------------------------- edge MLP + message scatter (WMMA)
// block = 128 threads (4 waves), 64 edges/block (16-row WMMA tile per wave)
// LDS: [W^T 320x320 bf16 = 200KB][edge_in 64x320 bf16 = 40KB][h 64x320 bf16 = 40KB]
__global__ __launch_bounds__(128) void k_edge(
        const bf16* __restrict__ snorm, const float* __restrict__ eattr,
        const float* __restrict__ evec,
        const bf16* __restrict__ w1t, const bf16* __restrict__ w2t,
        const float* __restrict__ eb1, const float* __restrict__ eb2,
        const long long* __restrict__ eidx, const float* __restrict__ vstate,
        float* __restrict__ s_accum, float* __restrict__ v_accum,
        float* __restrict__ counts) {
    extern __shared__ char smem[];
    bf16* lds_w   = (bf16*)smem;                  // 320*320
    bf16* lds_ein = (bf16*)(smem + 204800);       // 64*320
    bf16* lds_h   = (bf16*)(smem + 245760);       // 64*320
    const long long* tgt  = eidx;
    const long long* srcI = eidx + EE;
    int tid = threadIdx.x;
    long eBase = (long)blockIdx.x * 64;

    // stage W1^T into LDS with CDNA5 async-to-LDS (no VGPR round trip)
    for (int i = tid; i < (EHID * EIN * 2) / 16; i += 128)
        async_b128((char*)lds_w + i * 16, (const char*)w1t + i * 16);

    // prefetch W2^T (global_prefetch_b8) so it is L2-hot for the second stage
    for (int i = tid; i < (EOUT * EHID * 2) / 256; i += 128)
        __builtin_prefetch(((const char*)w2t) + (size_t)i * 256, 0, 1);

    // stage edge input rows: [snorm[tgt] | snorm[src] | bf16(edge_attr)]
    // gathered 16B chunks go straight to LDS via async; attr chunks need f32->bf16
    for (int idx = tid; idx < 64 * 40; idx += 128) {
        int row = idx / 40, c = idx % 40, f = c * 8;
        long e = eBase + row;
        bf16* dst = lds_ein + row * EIN + f;
        if (f < HH) {
            async_b128(dst, snorm + (long)tgt[e] * HH + f);
        } else if (f < 2 * HH) {
            async_b128(dst, snorm + (long)srcI[e] * HH + (f - HH));
        } else {
            const float* a = eattr + e * EDD + (f - 2 * HH);
            bf16x8 val;
            for (int j = 0; j < 8; j++) val[j] = (bf16)a[j];
            *(bf16x8*)dst = val;
        }
    }
    wait_async0();
    __syncthreads();

    int wave = tid >> 5, lane = tid & 31;
    int ln = lane & 15, half = lane >> 4;
    int rbase = wave * 16;

    // ---- layer 1: h = silu(edge_in @ W1 + b1), two n-tiles per K-loop ----
    {
        V16U af[10];
        load_afrag<10>(lds_ein + (rbase + ln) * EIN, half, af);
        for (int np = 0; np < 10; ++np) {
            v8f acc0 = {0.f, 0.f, 0.f, 0.f, 0.f, 0.f, 0.f, 0.f};
            v8f acc1 = {0.f, 0.f, 0.f, 0.f, 0.f, 0.f, 0.f, 0.f};
            int n0 = (2 * np) * 16 + ln, n1 = n0 + 16;
            wmma_pair<10>(af, lds_w + n0 * EIN, lds_w + n1 * EIN, half, acc0, acc1);
            float bias0 = eb1[n0], bias1 = eb1[n1];
            for (int r = 0; r < 8; r++) {
                int rr = rbase + r + 8 * half;
                float x0 = acc0[r] + bias0;
                float x1 = acc1[r] + bias1;
                lds_h[rr * EHID + n0] = (bf16)(x0 * sigmoidf_(x0));
                lds_h[rr * EHID + n1] = (bf16)(x1 * sigmoidf_(x1));
            }
        }
    }
    __syncthreads();
    // stage W2^T (overwrite W region) via async-to-LDS
    for (int i = tid; i < (EOUT * EHID * 2) / 16; i += 128)
        async_b128((char*)lds_w + i * 16, (const char*)w2t + i * 16);
    wait_async0();
    __syncthreads();

    V16U ah[10];
    load_afrag<10>(lds_h + (rbase + ln) * EHID, half, ah);

    // scalar message: sm * sigmoid(sg); tiles (t, t+8) pair on the same lane
    for (int t = 0; t < 8; ++t) {
        v8f am = {0.f, 0.f, 0.f, 0.f, 0.f, 0.f, 0.f, 0.f};
        v8f ag = {0.f, 0.f, 0.f, 0.f, 0.f, 0.f, 0.f, 0.f};
        int n = t * 16 + ln;
        wmma_pair<10>(ah, lds_w + n * EHID, lds_w + (n + HH) * EHID, half, am, ag);
        float b1v = eb2[n], b2v = eb2[n + HH];
        for (int r = 0; r < 8; r++) {
            long e = eBase + rbase + r + 8 * half;
            float val = (am[r] + b1v) * sigmoidf_(ag[r] + b2v);
            atomicAdd(s_accum + (long)tgt[e] * HH + n, val);
            if (t == 0 && ln == 0) atomicAdd(counts + tgt[e], 1.0f);
        }
    }
    // vector message: vstate[src]*sigmoid(vg) + unit*ds; tiles (16+p, 18+p)
    for (int p = 0; p < 2; p++) {
        v8f avg = {0.f, 0.f, 0.f, 0.f, 0.f, 0.f, 0.f, 0.f};
        v8f ads = {0.f, 0.f, 0.f, 0.f, 0.f, 0.f, 0.f, 0.f};
        int v = p * 16 + ln;
        int ng = 2 * HH + v, nd = 2 * HH + VV + v;
        wmma_pair<10>(ah, lds_w + ng * EHID, lds_w + nd * EHID, half, avg, ads);
        float bg = eb2[ng], bd = eb2[nd];
        for (int r = 0; r < 8; r++) {
            long e = eBase + rbase + r + 8 * half;
            float vgs = sigmoidf_(avg[r] + bg);
            float dsv = ads[r] + bd;
            float ex = evec[e * 3], ey = evec[e * 3 + 1], ez = evec[e * 3 + 2];
            float inv = 1.f / fmaxf(sqrtf(ex * ex + ey * ey + ez * ez), 1e-8f);
            float u[3] = {ex * inv, ey * inv, ez * inv};
            long s = (long)srcI[e], tg = (long)tgt[e];
            for (int d = 0; d < 3; d++) {
                float sv = vstate[(s * 3 + d) * VV + v];
                atomicAdd(v_accum + (tg * 3 + d) * VV + v, sv * vgs + u[d] * dsv);
            }
        }
    }
}

// ---------------- node residual, vnorm, 2nd layernorm, build node_in (bf16)
__global__ void k_node_stage(const float* __restrict__ sstate, const float* __restrict__ vstate,
                             const float* __restrict__ s_accum, const float* __restrict__ v_accum,
                             const float* __restrict__ counts,
                             const float* __restrict__ pg, const float* __restrict__ pb,
                             float* __restrict__ out_s, float* __restrict__ out_v,
                             bf16* __restrict__ node_in) {
    int wave = threadIdx.x >> 5, lane = threadIdx.x & 31;
    int row = blockIdx.x * 8 + wave;
    if (row >= NN) return;
    float rc = 1.f / fmaxf(counts[row], 1.f);
    float sv[4]; float sum = 0.f;
    for (int j = 0; j < 4; j++) {
        int f = lane + 32 * j;
        float x = sstate[(long)row * HH + f] + s_accum[(long)row * HH + f] * rc;
        out_s[(long)row * HH + f] = x;
        sv[j] = x; sum += x;
    }
    float sq = 0.f;
    for (int d = 0; d < 3; d++) {
        long o = ((long)row * 3 + d) * VV + lane;
        float x = vstate[o] + v_accum[o] * rc;
        out_v[o] = x; sq += x * x;
    }
    float vn = sqrtf(fmaxf(sq, 1e-8f));
    sum = wave_sum(sum);
    float m = sum * (1.f / HH);
    float var = 0.f;
    for (int j = 0; j < 4; j++) { float d = sv[j] - m; var += d * d; }
    var = wave_sum(var) * (1.f / HH);
    float inv = rsqrtf(var + 1e-5f);
    bf16* ni = node_in + (long)row * NIN;
    for (int j = 0; j < 4; j++) {
        int f = lane + 32 * j;
        ni[f] = (bf16)((sv[j] - m) * inv * pg[f] + pb[f]);
    }
    ni[HH + lane] = (bf16)vn;
}

// --------------------------------------------------- node MLP (WMMA) + update
// block = 128 threads (4 waves), 64 nodes/block
// LDS: [nW1^T 288x160][nW2^T 160x288][h2 64x288][A rows 64x160]
__global__ __launch_bounds__(128) void k_node_mlp(
        const bf16* __restrict__ node_in, const bf16* __restrict__ nw1t,
        const bf16* __restrict__ nw2t,
        const float* __restrict__ nb1, const float* __restrict__ nb2,
        float* __restrict__ out_s, float* __restrict__ out_v) {
    extern __shared__ char smem[];
    bf16* lds_w1 = (bf16*)smem;                   // 288*160
    bf16* lds_w2 = (bf16*)(smem + 92160);         // 160*288
    bf16* lds_h  = (bf16*)(smem + 184320);        // 64*288
    bf16* lds_a  = (bf16*)(smem + 221184);        // 64*160 staged node_in rows
    int tid = threadIdx.x;
    long rowBase0 = (long)blockIdx.x * 64;

    for (int i = tid; i < 92160 / 16; i += 128)
        async_b128((char*)lds_w1 + i * 16, (const char*)nw1t + i * 16);
    for (int i = tid; i < 92160 / 16; i += 128)
        async_b128((char*)lds_w2 + i * 16, (const char*)nw2t + i * 16);
    // stage this block's node_in rows (zero-padded past N)
    for (int idx = tid; idx < 64 * 20; idx += 128) {
        int row = idx / 20, c = idx % 20, f = c * 8;
        long gr = rowBase0 + row;
        bf16* dst = lds_a + row * NIN + f;
        if (gr < NN) {
            async_b128(dst, node_in + gr * NIN + f);
        } else {
            bf16x8 z;
            for (int j = 0; j < 8; j++) z[j] = (bf16)0.f;
            *(bf16x8*)dst = z;
        }
    }
    wait_async0();
    __syncthreads();

    int wave = tid >> 5, lane = tid & 31;
    int ln = lane & 15, half = lane >> 4;
    int rbase = wave * 16;
    long rowBase = rowBase0 + rbase;

    // layer 1: h2 = silu(node_in @ nW1 + nb1)   K=160 (5 ktiles), N=288 (18 ntiles)
    {
        V16U af[5];
        load_afrag<5>(lds_a + (rbase + ln) * NIN, half, af);
        for (int np = 0; np < 9; ++np) {
            v8f acc0 = {0.f, 0.f, 0.f, 0.f, 0.f, 0.f, 0.f, 0.f};
            v8f acc1 = {0.f, 0.f, 0.f, 0.f, 0.f, 0.f, 0.f, 0.f};
            int n0 = (2 * np) * 16 + ln, n1 = n0 + 16;
            wmma_pair<5>(af, lds_w1 + n0 * NIN, lds_w1 + n1 * NIN, half, acc0, acc1);
            float bias0 = nb1[n0], bias1 = nb1[n1];
            for (int r = 0; r < 8; r++) {
                int rr = rbase + r + 8 * half;
                float x0 = acc0[r] + bias0;
                float x1 = acc1[r] + bias1;
                lds_h[rr * NHID + n0] = (bf16)(x0 * sigmoidf_(x0));
                lds_h[rr * NHID + n1] = (bf16)(x1 * sigmoidf_(x1));
            }
        }
    }
    __syncthreads();

    // layer 2: nu = h2 @ nW2 + nb2   K=288 (9 ktiles), N=160 (10 ntiles)
    V16U ah[9];
    load_afrag<9>(lds_h + (rbase + ln) * NHID, half, ah);
    for (int tp = 0; tp < 5; ++tp) {
        v8f acc0 = {0.f, 0.f, 0.f, 0.f, 0.f, 0.f, 0.f, 0.f};
        v8f acc1 = {0.f, 0.f, 0.f, 0.f, 0.f, 0.f, 0.f, 0.f};
        int t0 = 2 * tp, t1 = 2 * tp + 1;
        int n0 = t0 * 16 + ln, n1 = t1 * 16 + ln;
        wmma_pair<9>(ah, lds_w2 + n0 * NHID, lds_w2 + n1 * NHID, half, acc0, acc1);
        for (int which = 0; which < 2; which++) {
            int n = which ? n1 : n0;
            int t = which ? t1 : t0;
            v8f& acc = which ? acc1 : acc0;
            float bias = nb2[n];
            for (int r = 0; r < 8; r++) {
                long row = rowBase + r + 8 * half;
                if (row >= NN) continue;
                float x = acc[r] + bias;
                if (t < 8) {
                    out_s[row * HH + n] += x;                 // scalar_delta
                } else {
                    int v = n - HH;
                    float sc = 1.f + 0.1f * tanhf(x);         // vector_scale
                    for (int d = 0; d < 3; d++) out_v[(row * 3 + d) * VV + v] *= sc;
                }
            }
        }
    }
}

extern "C" void kernel_launch(void* const* d_in, const int* in_sizes, int n_in,
                              void* d_out, int out_size, void* d_ws, size_t ws_size,
                              hipStream_t stream) {
    const float* sstate = (const float*)d_in[0];
    const float* vstate = (const float*)d_in[1];
    const float* eattr  = (const float*)d_in[2];
    const float* evec   = (const float*)d_in[3];
    const float* sng    = (const float*)d_in[4];
    const float* snb    = (const float*)d_in[5];
    const float* png    = (const float*)d_in[6];
    const float* pnb    = (const float*)d_in[7];
    const float* ew1    = (const float*)d_in[8];
    const float* eb1    = (const float*)d_in[9];
    const float* ew2    = (const float*)d_in[10];
    const float* eb2    = (const float*)d_in[11];
    const float* nw1    = (const float*)d_in[12];
    const float* nb1    = (const float*)d_in[13];
    const float* nw2    = (const float*)d_in[14];
    const float* nb2    = (const float*)d_in[15];
    const long long* eidx = (const long long*)d_in[16];

    float* out_s = (float*)d_out;
    float* out_v = out_s + (long)NN * HH;

    char* ws = (char*)d_ws;
    size_t off = 0;
    auto alloc = [&](size_t bytes) -> void* {
        void* p = ws + off;
        off = (off + bytes + 255) & ~(size_t)255;
        return p;
    };
    bf16*  snorm   = (bf16*) alloc((size_t)NN * HH * 2);
    bf16*  w1t     = (bf16*) alloc((size_t)EHID * EIN * 2);
    bf16*  w2t     = (bf16*) alloc((size_t)EOUT * EHID * 2);
    bf16*  nw1t    = (bf16*) alloc((size_t)NHID * NIN * 2);
    bf16*  nw2t    = (bf16*) alloc((size_t)NOUT * NHID * 2);
    float* s_accum = (float*)alloc((size_t)NN * HH * 4);
    float* v_accum = (float*)alloc((size_t)NN * 3 * VV * 4);
    float* counts  = (float*)alloc((size_t)NN * 4);
    bf16*  node_in = (bf16*) alloc((size_t)NN * NIN * 2);

    k_zero<<<2048, 256, 0, stream>>>(s_accum, (long)NN * HH);
    k_zero<<<2048, 256, 0, stream>>>(v_accum, (long)NN * 3 * VV);
    k_zero<<<256,  256, 0, stream>>>(counts,  (long)NN);

    k_ln_cast<<<(NN + 7) / 8, 256, 0, stream>>>(sstate, sng, snb, snorm);
    k_convert<<<512, 256, 0, stream>>>(ew1, ew2, nw1, nw2, w1t, w2t, nw1t, nw2t);

    // 280KB dynamic LDS per workgroup (W^T 200KB + edge_in 40KB + h 40KB)
    k_edge<<<EE / 64, 128, 286720, stream>>>(snorm, eattr, evec, w1t, w2t, eb1, eb2,
                                             eidx, vstate, s_accum, v_accum, counts);

    k_node_stage<<<(NN + 7) / 8, 256, 0, stream>>>(sstate, vstate, s_accum, v_accum,
                                                   counts, png, pnb, out_s, out_v, node_in);

    // 236KB dynamic LDS per workgroup (weights 180KB + h2 36KB + A rows 20KB)
    k_node_mlp<<<(NN + 63) / 64, 128, 241664, stream>>>(node_in, nw1t, nw2t, nb1, nb2,
                                                        out_s, out_v);
}
